// GaborRenderer_v55_53420803228222
// MI455X (gfx1250) — matrix operand order repaired and checked by reading, not compiled, not run on hardware
//
#include <hip/hip_runtime.h>
#include <hip/hip_bf16.h>
#include <math.h>

// ---------------------------------------------------------------------------
// GaborRenderer fused pipeline for gfx1250 (MI455X), wave32 + WMMA f16 +
// async-to-LDS weight streaming (ASYNCcnt software pipeline, 2 chunks deep).
//
// Shapes: B=8, N=16384, M=64, D=H=256, 8 heads x 32, OUT folded to 32 (t,c).
// 256-thread blocks = 8 waves (2 per SIMD32) sharing the staged weight tiles;
// each wave owns a 16-point row tile -> 128 points per workgroup.
// All K=256 GEMMs run as v_wmma_f32_16x16x32_f16: 16 C-tiles x 8 k-tiles.
// Weight matrices are split into two 128-row halves; halves are streamed into
// two alternating LDS buffers with global_load_async_to_lds_b128 and consumed
// under s_wait_asynccnt thresholds so WMMA compute overlaps weight traffic.
// ---------------------------------------------------------------------------

typedef __attribute__((ext_vector_type(16))) _Float16 v16h;
typedef __attribute__((ext_vector_type(8)))  _Float16 v8h;
typedef __attribute__((ext_vector_type(8)))  float    v8f;

union V16 { v16h v; v8h h[2]; };

#define NPTS      16384
#define NTHREADS  256
#define NWAVES    8
#define ROWS_WG   128           // 8 waves * 16 rows
#define XSTRIDE   264           // halves; 528B rows -> bank-conflict-free frag loads
#define SSTRIDE   80            // halves; attention scores buffer stride
#define HBYTES    67584         // 128 rows * 528B : one half-matrix LDS buffer
#define WBYTES    (2*HBYTES)    // two alternating half buffers
#define WAVEBYTES 19456         // xbuf(8448) + zbuf(8448) + sbuf(2560)
#define SMEM_TOTAL (WBYTES + NWAVES*WAVEBYTES)   // 290816 <= 320KB
#define NCHUNKS   24            // 12 matrices * 2 halves
#define LDS_PER_CHUNK 16        // async b128 loads per thread per chunk

// ---------------- fragment helpers (contiguous k order, same for A and B) ---------

__device__ __forceinline__ v16h ldA(const _Float16* buf, int stride, int lane, int kt) {
  // A 16x32 f16: lane -> M=lane&15 ; elem j -> K = 16*(lane>>4) + j (contiguous)
  int row = lane & 15;
  int kb  = kt * 32 + ((lane >> 4) << 4);
  V16 r;
  r.h[0] = *(const v8h*)(buf + row * stride + kb);
  r.h[1] = *(const v8h*)(buf + row * stride + kb + 8);
  return r.v;
}

__device__ __forceinline__ v16h ldB(const _Float16* w, int stride, int lane, int ot, int kt) {
  // B 32x16 f16: lane -> N=lane&15 ; elem j -> K = 16*(lane>>4) + j (contiguous)
  int o = ot * 16 + (lane & 15);
  int k = kt * 32 + ((lane >> 4) << 4);
  V16 r;
  r.h[0] = *(const v8h*)(w + o * stride + k);
  r.h[1] = *(const v8h*)(w + o * stride + k + 8);
  return r.v;
}

__device__ __forceinline__ v16h ldBglobal(const _Float16* p) {
  V16 r;
  r.h[0] = *(const v8h*)(p);
  r.h[1] = *(const v8h*)(p + 8);
  return r.v;
}

__device__ __forceinline__ v8f wmma_f16(v16h a, v16h b, v8f c) {
  return __builtin_amdgcn_wmma_f32_16x16x32_f16(false, a, false, b, (short)0, c, false, false);
}

__device__ __forceinline__ v8f zero8() {
  v8f z;
#pragma unroll
  for (int i = 0; i < 8; ++i) z[i] = 0.0f;
  return z;
}

// ---------------- register-tile helpers -------------------------------------

__device__ __forceinline__ void biasAcc(v8f acc[16], const float* bias, int lane) {
#pragma unroll
  for (int ot = 0; ot < 16; ++ot) {
    float bv = bias[ot * 16 + (lane & 15)];
#pragma unroll
    for (int vi = 0; vi < 8; ++vi) acc[ot][vi] = bv;
  }
}

__device__ __forceinline__ void zeroAcc(v8f acc[16]) {
#pragma unroll
  for (int ot = 0; ot < 16; ++ot) acc[ot] = zero8();
}

__device__ __forceinline__ void scaleAcc(v8f acc[16], float s) {
#pragma unroll
  for (int ot = 0; ot < 16; ++ot)
#pragma unroll
    for (int vi = 0; vi < 8; ++vi) acc[ot][vi] *= s;
}

// half GEMM: acc[OB..OB+7] += A(abuf) x W_half^T (128 rows staged in LDS half)
template <int OB>
__device__ __forceinline__ void gemm8(v8f acc[16], const _Float16* w,
                                      const _Float16* abuf, int lane) {
#pragma unroll 1
  for (int kt = 0; kt < 8; ++kt) {
    v16h a = ldA(abuf, XSTRIDE, lane, kt);
#pragma unroll
    for (int ot = 0; ot < 8; ++ot) {
      v16h bf = ldB(w, XSTRIDE, lane, ot, kt);
      acc[OB + ot] = wmma_f16(a, bf, acc[OB + ot]);
    }
  }
}

__device__ __forceinline__ void storeAcc(_Float16* buf, const v8f acc[16], int lane) {
  int rb = (lane >> 4) << 3;
  int c0 = lane & 15;
#pragma unroll
  for (int ot = 0; ot < 16; ++ot)
#pragma unroll
    for (int vi = 0; vi < 8; ++vi)
      buf[(vi + rb) * XSTRIDE + ot * 16 + c0] = (_Float16)acc[ot][vi];
}

__device__ __forceinline__ float red16(float x) {
  x += __shfl_xor(x, 1, 32);
  x += __shfl_xor(x, 2, 32);
  x += __shfl_xor(x, 4, 32);
  x += __shfl_xor(x, 8, 32);
  return x;
}

__device__ __forceinline__ float redmax16(float x) {
  x = fmaxf(x, __shfl_xor(x, 1, 32));
  x = fmaxf(x, __shfl_xor(x, 2, 32));
  x = fmaxf(x, __shfl_xor(x, 4, 32));
  x = fmaxf(x, __shfl_xor(x, 8, 32));
  return x;
}

__device__ __forceinline__ void layernormAcc(v8f acc[16], const float* g,
                                             const float* bt, int lane) {
  float mus[8], rss[8];
#pragma unroll
  for (int vi = 0; vi < 8; ++vi) {
    float s = 0.f, s2 = 0.f;
#pragma unroll
    for (int ot = 0; ot < 16; ++ot) { float v = acc[ot][vi]; s += v; s2 += v * v; }
    s = red16(s); s2 = red16(s2);
    float mu  = s  * (1.0f / 256.0f);
    float var = s2 * (1.0f / 256.0f) - mu * mu;
    mus[vi] = mu;
    rss[vi] = rsqrtf(var + 1e-5f);
  }
#pragma unroll
  for (int ot = 0; ot < 16; ++ot) {
    float gg = g [ot * 16 + (lane & 15)];
    float bb = bt[ot * 16 + (lane & 15)];
#pragma unroll
    for (int vi = 0; vi < 8; ++vi)
      acc[ot][vi] = (acc[ot][vi] - mus[vi]) * rss[vi] * gg + bb;
  }
}

// multiply acc elementwise by Gabor basis of filter `filt`
__device__ __forceinline__ void gaborMul(v8f acc[16], int filt,
                                         const float* gW, const float* gB,
                                         const float* gMu, const float* gG,
                                         const float cx[8], const float cy[8],
                                         const float cn[8], int lane) {
  int c0 = lane & 15;
#pragma unroll
  for (int ot = 0; ot < 16; ++ot) {
    int hc   = filt * 256 + ot * 16 + c0;
    float w0 = gW [hc * 2 + 0];
    float w1 = gW [hc * 2 + 1];
    float bb = gB [hc];
    float m0 = gMu[hc * 2 + 0];
    float m1 = gMu[hc * 2 + 1];
    float gm = gG [hc];
    float muu = m0 * m0 + m1 * m1;
#pragma unroll
    for (int vi = 0; vi < 8; ++vi) {
      float D = cn[vi] + muu - 2.f * (cx[vi] * m0 + cy[vi] * m1);
      float s = __sinf(cx[vi] * w0 + cy[vi] * w1 + bb) * __expf(-0.5f * D * gm);
      acc[ot][vi] *= s;
    }
  }
}

// ---------------- async weight streaming (ASYNCcnt pipeline) ----------------
// chunk c (0..23): matrix use-index u=c>>1, half hf=c&1.
// half hf of a matrix (128 rows x 512B) lands in LDS buffer hf at base hf*HBYTES.
// Each thread issues 16 global_load_async_to_lds_b128 (16B each).

__device__ __forceinline__ int slotOf(int u) {
  if (u <  3) return u;            // qp_W2, Wq, out_W
  if (u == 3) return 7;            // net2_0
  return (u & 1) ? (8 + ((u - 5) >> 1))   // net2_1..4 -> slots 8..11
                 : (3 + ((u - 4) >> 1));  // net1_0..3 -> slots 3..6
}

__device__ __forceinline__ void issueChunk(int c, const _Float16* wslots, int tid) {
  if (c >= NCHUNKS) return;
  int u = c >> 1, hf = c & 1;
  const _Float16* src = wslots + slotOf(u) * 65536 + hf * 32768;
  unsigned lbase = hf ? (unsigned)HBYTES : 0u;
#pragma unroll 1
  for (int j = 0; j < LDS_PER_CHUNK; ++j) {
    int i = tid + j * NTHREADS;                              // 16B unit index
    unsigned lds  = lbase + (unsigned)(i >> 5) * 528u + (unsigned)(i & 31) * 16u;
    unsigned goff = (unsigned)i * 16u;
    asm volatile("global_load_async_to_lds_b128 %0, %1, %2"
                 :: "v"(lds), "v"(goff), "s"(src) : "memory");
  }
}

// wait for the oldest in-flight chunk; `more`=true if one newer chunk
// (LDS_PER_CHUNK loads) should remain outstanding. Async loads complete in
// order, so a threshold of 16 retires exactly the older chunk.
__device__ __forceinline__ void waitChunk(bool more) {
  if (more) asm volatile("s_wait_asynccnt 0x10" ::: "memory");
  else      asm volatile("s_wait_asynccnt 0x0"  ::: "memory");
  __syncthreads();
}

// consume one full 256x256 matrix (chunks CIDX, CIDX+1), prefetch CIDX+2/3
#define CONSUME_MATRIX(ABUF, CIDX)                                  \
  do {                                                              \
    waitChunk((CIDX) + 1 < NCHUNKS);                                \
    gemm8<0>(acc, wbuf0, (ABUF), lane);                             \
    __syncthreads();                                                \
    issueChunk((CIDX) + 2, wslots, tid);                            \
    waitChunk((CIDX) + 2 < NCHUNKS);                                \
    gemm8<8>(acc, wbuf1, (ABUF), lane);                             \
    __syncthreads();                                                \
    issueChunk((CIDX) + 3, wslots, tid);                            \
  } while (0)

// ---------------------------------------------------------------------------
// prep kernels
// ---------------------------------------------------------------------------

// convert 12 [256][256] f32 weight matrices to f16 slots (row-major, unpadded)
__global__ void prep_weights(const float* qpW2, const float* inW, const float* outW,
                             const float* net1W, const float* net2W, _Float16* dst) {
  int e = blockIdx.x * 256 + threadIdx.x;   // 12*65536 total
  int slot = e >> 16, r = e & 65535;
  float v;
  if      (slot == 0) v = qpW2[r];
  else if (slot == 1) v = inW[r];                        // Wq = in_W rows 0..255
  else if (slot == 2) v = outW[r];
  else if (slot <  7) v = net1W[(slot - 3) * 65536 + r];
  else                v = net2W[(slot - 7) * 65536 + r];
  dst[e] = (_Float16)v;
}

// fold the ortho irfft into fin_W: W2[32][256] = G @ fin_W, b2[32] = G @ fin_b
__global__ void prep_fin2(const float* finW, const float* finb,
                          _Float16* w2, float* b2) {
  int o2 = blockIdx.x;            // (t,c): o2 = t*2 + c
  int k  = threadIdx.x;           // 0..255
  int t = o2 >> 1, c = o2 & 1;
  float acc = 0.f, bacc = 0.f;
  for (int j = 0; j < 36; ++j) {
    int f = j >> 2, cc = (j >> 1) & 1, ri = j & 1;
    float g = 0.f;
    if (cc == c) {
      float wgt = (f == 0 || f == 8) ? 0.25f : 0.5f;     // 1/sqrt(16) * {1,2}
      float ang = 6.2831853071795864f * (float)(f * t) / 16.f;
      if (ri == 0)                       g =  wgt * cosf(ang);
      else if (f != 0 && f != 8)         g = -wgt * sinf(ang);  // imag of DC/Nyq ignored
    }
    acc  += g * finW[j * 256 + k];
    bacc += g * finb[j];
  }
  w2[o2 * 256 + k] = (_Float16)acc;
  if (k == 0) b2[o2] = bacc;
}

// layernorm(z_multi) then K,V projections; store f16 in fragment-friendly layout
// Kf[b][h][m][32d], Vf[b][h][32d][64m]
__global__ void __launch_bounds__(256) prep_kv(const float* z, const float* inW,
                                               const float* inb, const float* g,
                                               const float* bt,
                                               _Float16* Kf, _Float16* Vf) {
  __shared__ float row[256];
  __shared__ float red[256];
  int bm = blockIdx.x;            // 0..511 = b*64+m
  int bb = bm >> 6, m = bm & 63;
  int t  = threadIdx.x;
  float v = z[(bb * 64 + m) * 256 + t];
  red[t] = v; __syncthreads();
  for (int s = 128; s > 0; s >>= 1) { if (t < s) red[t] += red[t + s]; __syncthreads(); }
  float mu = red[0] / 256.f; __syncthreads();
  float d = v - mu;
  red[t] = d * d; __syncthreads();
  for (int s = 128; s > 0; s >>= 1) { if (t < s) red[t] += red[t + s]; __syncthreads(); }
  float rs = rsqrtf(red[0] / 256.f + 1e-5f);
  row[t] = d * rs * g[t] + bt[t];
  __syncthreads();
  float ka = inb[256 + t], va = inb[512 + t];
  for (int dd = 0; dd < 256; ++dd) {
    float rv = row[dd];
    ka += rv * inW[(256 + t) * 256 + dd];
    va += rv * inW[(512 + t) * 256 + dd];
  }
  int hh = t >> 5, di = t & 31;
  Kf[(((bb * 8 + hh) * 64) + m) * 32 + di] = (_Float16)ka;
  Vf[(((bb * 8 + hh) * 32) + di) * 64 + m] = (_Float16)va;
}

// ---------------------------------------------------------------------------
// main fused kernel
// ---------------------------------------------------------------------------

__global__ void __launch_bounds__(NTHREADS) gabor_main(
    const float* __restrict__ coords,
    const float* __restrict__ gW, const float* __restrict__ gB,
    const float* __restrict__ gMu, const float* __restrict__ gG,
    const float* __restrict__ net1_b,
    const float* __restrict__ qpW1, const float* __restrict__ qpb1,
    const float* __restrict__ qpb2,
    const float* __restrict__ in_b, const float* __restrict__ out_b,
    const float* __restrict__ lnq_g, const float* __restrict__ lnq_b,
    const _Float16* __restrict__ wslots,
    const _Float16* __restrict__ fin2W, const float* __restrict__ fin2b,
    const _Float16* __restrict__ Kf, const _Float16* __restrict__ Vf,
    float* __restrict__ out) {
  extern __shared__ char smem[];
  _Float16* wbuf0 = (_Float16*)smem;                  // half-matrix buffer 0
  _Float16* wbuf1 = (_Float16*)(smem + HBYTES);       // half-matrix buffer 1
  int tid  = threadIdx.x;
  int wave = tid >> 5;
  int lane = tid & 31;
  _Float16* xbuf = (_Float16*)(smem + WBYTES + wave * WAVEBYTES);
  _Float16* zbuf = xbuf + 16 * XSTRIDE;
  _Float16* sbuf = zbuf + 16 * XSTRIDE;

  const int nb = NPTS / ROWS_WG;             // 128 blocks per batch
  int b       = blockIdx.x / nb;
  int rowbase = (blockIdx.x % nb) * ROWS_WG + wave * 16;

  // kick off the async weight pipeline (overlaps with the q1 prologue)
  issueChunk(0, wslots, tid);
  issueChunk(1, wslots, tid);

  // per-lane coordinates for the 8 rows this lane-half covers
  float cx[8], cy[8], cn[8];
#pragma unroll
  for (int vi = 0; vi < 8; ++vi) {
    int r = vi + ((lane >> 4) << 3);
    int n = rowbase + r;
    float x = coords[(b * NPTS + n) * 2 + 0];
    float y = coords[(b * NPTS + n) * 2 + 1];
    cx[vi] = x; cy[vi] = y; cn[vi] = x * x + y * y;
  }

  // ---- layer 0: q1 = GELU(x0 @ qpW1^T + b1) -> xbuf (f16, row-major) ------
#pragma unroll 1
  for (int r = 0; r < 16; ++r) {
    int n = rowbase + r;
    float x = coords[(b * NPTS + n) * 2 + 0];
    float y = coords[(b * NPTS + n) * 2 + 1];
#pragma unroll 1
    for (int t = 0; t < 8; ++t) {
      int hcol = t * 32 + lane;
      float v = qpW1[hcol * 2 + 0] * x + qpW1[hcol * 2 + 1] * y + qpb1[hcol];
      v = 0.5f * v * (1.f + erff(v * 0.70710678118f));
      xbuf[r * XSTRIDE + hcol] = (_Float16)v;
    }
  }

  v8f acc[16];

  // ---- q2 = q1 @ qpW2^T + b2, layernorm ----------------------------------- (chunks 0,1)
  biasAcc(acc, qpb2, lane);
  CONSUME_MATRIX(xbuf, 0);
  layernormAcc(acc, lnq_g, lnq_b, lane);
  storeAcc(xbuf, acc, lane);

  // ---- Q = q_ln @ Wq^T + bq, pre-scaled by 1/sqrt(32) --------------------- (chunks 2,3)
  biasAcc(acc, in_b, lane);
  CONSUME_MATRIX(xbuf, 2);
  scaleAcc(acc, 0.17677669529f);
  storeAcc(xbuf, acc, lane);

  // ---- attention (overlaps with out_W streaming) -----------------
#pragma unroll 1
  for (int hd = 0; hd < 8; ++hd) {
    v16h aq = ldA(xbuf, XSTRIDE, lane, hd);       // Q head tile 16x32
    v8f sc[4];
#pragma unroll
    for (int mt = 0; mt < 4; ++mt) {
      const _Float16* kp = Kf + ((((b * 8 + hd) * 64) + mt * 16 + (lane & 15)) * 32)
                              + ((lane >> 4) << 4);
      sc[mt] = wmma_f16(aq, ldBglobal(kp), zero8());
    }
    // row-wise softmax over 64 keys
#pragma unroll
    for (int vi = 0; vi < 8; ++vi) {
      float mx = -1e30f;
#pragma unroll
      for (int mt = 0; mt < 4; ++mt) mx = fmaxf(mx, sc[mt][vi]);
      mx = redmax16(mx);
      float s = 0.f;
#pragma unroll
      for (int mt = 0; mt < 4; ++mt) { float e = __expf(sc[mt][vi] - mx); sc[mt][vi] = e; s += e; }
      s = red16(s);
      float inv = 1.f / s;
#pragma unroll
      for (int mt = 0; mt < 4; ++mt) sc[mt][vi] *= inv;
    }
    // attn -> sbuf [16 x 64] f16
    {
      int rb = (lane >> 4) << 3, c0 = lane & 15;
#pragma unroll
      for (int mt = 0; mt < 4; ++mt)
#pragma unroll
        for (int vi = 0; vi < 8; ++vi)
          sbuf[(vi + rb) * SSTRIDE + mt * 16 + c0] = (_Float16)sc[mt][vi];
    }
    // ctx = attn @ V -> zbuf cols [hd*32 .. hd*32+31]
#pragma unroll
    for (int dt = 0; dt < 2; ++dt) {
      v8f cacc = zero8();
#pragma unroll
      for (int kt = 0; kt < 2; ++kt) {
        v16h a = ldA(sbuf, SSTRIDE, lane, kt);
        const _Float16* vp = Vf + ((((b * 8 + hd) * 32) + dt * 16 + (lane & 15)) * 64)
                                + kt * 32 + ((lane >> 4) << 4);
        cacc = wmma_f16(a, ldBglobal(vp), cacc);
      }
      int rb = (lane >> 4) << 3, c0 = lane & 15;
#pragma unroll
      for (int vi = 0; vi < 8; ++vi)
        zbuf[(vi + rb) * XSTRIDE + hd * 32 + dt * 16 + c0] = (_Float16)cacc[vi];
    }
  }

  // ---- z = ctx @ outW^T + out_b -> zbuf ----------------------------------- (chunks 4,5)
  biasAcc(acc, out_b, lane);
  CONSUME_MATRIX(zbuf, 4);
  storeAcc(zbuf, acc, lane);

  // ---- MFN layer 0: x = gabor0 * (z @ net2_0^T) --------------------------- (chunks 6,7)
  zeroAcc(acc);
  CONSUME_MATRIX(zbuf, 6);
  gaborMul(acc, 0, gW, gB, gMu, gG, cx, cy, cn, lane);
  storeAcc(xbuf, acc, lane);

  // ---- MFN layers 1..4 ---------------------------------------------------- (chunks 8..23)
#pragma unroll 1
  for (int i = 1; i <= 4; ++i) {
    int c1 = 8 + (i - 1) * 4;          // net1_{i-1}
    int c2 = c1 + 2;                   // net2_i
    biasAcc(acc, net1_b + (i - 1) * 256, lane);
    CONSUME_MATRIX(xbuf, c1);
    CONSUME_MATRIX(zbuf, c2);
    gaborMul(acc, i, gW, gB, gMu, gG, cx, cy, cn, lane);
    storeAcc(xbuf, acc, lane);
  }

  // ---- final: y = x @ (G*fin_W)^T + G*fin_b  (irfft folded in) ------------
  v8f f0, f1;
  {
    float b0 = fin2b[(lane & 15)];
    float b1 = fin2b[16 + (lane & 15)];
#pragma unroll
    for (int vi = 0; vi < 8; ++vi) { f0[vi] = b0; f1[vi] = b1; }
  }
#pragma unroll 1
  for (int kt = 0; kt < 8; ++kt) {
    v16h a = ldA(xbuf, XSTRIDE, lane, kt);
    const _Float16* p0 = fin2W + ((lane & 15)) * 256 + kt * 32 + ((lane >> 4) << 4);
    const _Float16* p1 = fin2W + (16 + (lane & 15)) * 256 + kt * 32 + ((lane >> 4) << 4);
    f0 = wmma_f16(a, ldBglobal(p0), f0);
    f1 = wmma_f16(a, ldBglobal(p1), f1);
  }
  // scatter to out[B, 16, N, 2]
#pragma unroll
  for (int vi = 0; vi < 8; ++vi) {
    int r = vi + ((lane >> 4) << 3);
    int n = rowbase + r;
    int c = lane & 15;
    { int tt = c >> 1, ch = c & 1;
      out[((b * 16 + tt) * NPTS + n) * 2 + ch] = f0[vi]; }
    { int c2 = 16 + c; int tt = c2 >> 1, ch = c2 & 1;
      out[((b * 16 + tt) * NPTS + n) * 2 + ch] = f1[vi]; }
  }
}

// ---------------------------------------------------------------------------
// launch
// ---------------------------------------------------------------------------

extern "C" void kernel_launch(void* const* d_in, const int* in_sizes, int n_in,
                              void* d_out, int out_size, void* d_ws, size_t ws_size,
                              hipStream_t stream) {
  const float* z_multi = (const float*)d_in[0];
  const float* coords  = (const float*)d_in[1];
  const float* gabor_W = (const float*)d_in[2];
  const float* gabor_b = (const float*)d_in[3];
  const float* gabor_mu= (const float*)d_in[4];
  const float* gabor_g = (const float*)d_in[5];
  const float* net1_W  = (const float*)d_in[6];
  const float* net1_b  = (const float*)d_in[7];
  const float* net2_W  = (const float*)d_in[8];
  const float* qp_W1   = (const float*)d_in[9];
  const float* qp_b1   = (const float*)d_in[10];
  const float* qp_W2   = (const float*)d_in[11];
  const float* qp_b2   = (const float*)d_in[12];
  const float* in_W    = (const float*)d_in[13];
  const float* in_b    = (const float*)d_in[14];
  const float* out_W   = (const float*)d_in[15];
  const float* out_b   = (const float*)d_in[16];
  const float* lnq_g   = (const float*)d_in[17];
  const float* lnq_b   = (const float*)d_in[18];
  const float* lnkv_g  = (const float*)d_in[19];
  const float* lnkv_b  = (const float*)d_in[20];
  const float* fin_W   = (const float*)d_in[21];
  const float* fin_b   = (const float*)d_in[22];
  (void)in_sizes; (void)n_in; (void)out_size; (void)ws_size;

  char* ws = (char*)d_ws;
  _Float16* wslots = (_Float16*)(ws);                 // 12 slots * 128KB
  _Float16* fin2W  = (_Float16*)(ws + 1572864);       // 16KB
  float*    fin2b  = (float*)   (ws + 1589248);       // 128B
  _Float16* Kf     = (_Float16*)(ws + 1589376);       // 256KB
  _Float16* Vf     = (_Float16*)(ws + 1851520);       // 256KB

  prep_weights<<<3072, 256, 0, stream>>>(qp_W2, in_W, out_W, net1_W, net2_W, wslots);
  prep_fin2<<<32, 256, 0, stream>>>(fin_W, fin_b, fin2W, fin2b);
  prep_kv<<<512, 256, 0, stream>>>(z_multi, in_W, in_b, lnkv_g, lnkv_b, Kf, Vf);

  hipFuncSetAttribute((const void*)gabor_main,
                      hipFuncAttributeMaxDynamicSharedMemorySize, SMEM_TOTAL);
  gabor_main<<<NPTS / ROWS_WG * 8, NTHREADS, SMEM_TOTAL, stream>>>(
      coords, gabor_W, gabor_b, gabor_mu, gabor_g,
      net1_b, qp_W1, qp_b1, qp_b2, in_b, out_b, lnq_g, lnq_b,
      wslots, fin2W, fin2b, Kf, Vf, (float*)d_out);
}
